// DensePoseV1ConvXGNSparseGNHead_36807869727080
// MI455X (gfx1250) — matrix operand order repaired and checked by reading, not compiled, not run on hardware
//
#include <hip/hip_runtime.h>
#include <hip/hip_bf16.h>

typedef __attribute__((ext_vector_type(16))) _Float16 v16h;
typedef __attribute__((ext_vector_type(8)))  float    v8f;

#define N_PTS   500000
#define C_CH    64
#define NUM_INST 64
#define NGROUP  32
#define EPS_GN  1e-05f

// ---------------------------------------------------------------------------
// zero a float buffer
__global__ void k_zero_f(float* __restrict__ p, int n) {
    int i = blockIdx.x * 256 + threadIdx.x;
    if (i < n) p[i] = 0.0f;
}

// ---------------------------------------------------------------------------
// per-instance point count (LDS histogram -> global float atomics)
__global__ void k_cnt_hist(const int* __restrict__ ins, float* __restrict__ cnt, int n) {
    __shared__ int bins[NUM_INST];
    if (threadIdx.x < NUM_INST) bins[threadIdx.x] = 0;
    __syncthreads();
    int i = blockIdx.x * 256 + threadIdx.x;
    if (i < n) atomicAdd(&bins[ins[i]], 1);
    __syncthreads();
    if (threadIdx.x < NUM_INST) {
        int b = bins[threadIdx.x];
        if (b) atomicAdd(&cnt[threadIdx.x], (float)b);
    }
}

// ---------------------------------------------------------------------------
// features f32 -> f16 (N+1 rows; row N is zeros for masked gather)
__global__ void k_cvt_x(const float* __restrict__ f, _Float16* __restrict__ xh,
                        int total, int nvalid) {
    int i = blockIdx.x * 256 + threadIdx.x;
    if (i < total) xh[i] = (i < nvalid) ? (_Float16)f[i] : (_Float16)0.0f;
}

// ---------------------------------------------------------------------------
// W[9][cin][cout] f32 -> Wt[9][cout][cin] f16 (B-fragment-friendly)
__global__ void k_cvt_w(const float* __restrict__ W, _Float16* __restrict__ Wt) {
    int i = blockIdx.x * 256 + threadIdx.x;            // 9*64*64 = 36864
    if (i < 9 * 64 * 64) {
        int k  = i >> 12;
        int ci = (i >> 6) & 63;
        int co = i & 63;
        Wt[(k << 12) + (co << 6) + ci] = (_Float16)W[i];
    }
}

// ---------------------------------------------------------------------------
// Sparse 9-tap conv via WMMA: one wave per PAIR of 16-point tiles (32 points),
// full 64 out-channels. Each B-fragment load feeds two WMMAs (M-tile reuse).
// A (16x32 f16)  : lane L -> row M = L%16; chunks K = 32s+{0|8}, 32s+16+{0|8}
// B (32x16 f16)  : lane L -> col N = L%16; chunks K = 32s+{0|8}, 32s+16+{0|8}
// C/D (16x16 f32): VGPR v, lane L -> M = v + (L>=16?8:0), N = L%16
__global__ __launch_bounds__(256)
void k_conv_wmma(const _Float16* __restrict__ Xh,   // (N+1) x 64
                 const _Float16* __restrict__ Wt,   // 9 x 64(co) x 64(ci)
                 const int*      __restrict__ nbr,  // N x 9
                 const float*    __restrict__ bias, // 64
                 float*          __restrict__ Y,    // N x 64
                 int npairs, int zero_row) {
    const int lane = threadIdx.x & 31;
    const int wave = threadIdx.x >> 5;
    const int pair = blockIdx.x * 8 + wave;
    if (pair >= npairs) return;                       // wave-uniform: EXEC stays full

    const int r   = lane & 15;
    const int hiA = (lane & 16) ? 8 : 0;              // A half-wave K offset
    const int hiB = (lane & 16) ? 8 : 0;              // B half-wave K offset
    const int pA  = pair * 32 + r;                    // lane's row, tile A
    const int pB  = pA + 16;                          // lane's row, tile B

    v8f accA0, accA1, accA2, accA3, accB0, accB1, accB2, accB3;
    {
        float b0 = bias[r], b1 = bias[16 + r], b2 = bias[32 + r], b3 = bias[48 + r];
        accA0 = (v8f){b0, b0, b0, b0, b0, b0, b0, b0};
        accA1 = (v8f){b1, b1, b1, b1, b1, b1, b1, b1};
        accA2 = (v8f){b2, b2, b2, b2, b2, b2, b2, b2};
        accA3 = (v8f){b3, b3, b3, b3, b3, b3, b3, b3};
        accB0 = accA0; accB1 = accA1; accB2 = accA2; accB3 = accA3;
    }

    for (int k = 0; k < 9; ++k) {
        int idxA = nbr[pA * 9 + k];
        int idxB = nbr[pB * 9 + k];
        const _Float16* arowA = Xh + (size_t)(idxA < 0 ? zero_row : idxA) * 64;
        const _Float16* arowB = Xh + (size_t)(idxB < 0 ? zero_row : idxB) * 64;
        const _Float16* wk    = Wt + (size_t)k * 4096;
#pragma unroll
        for (int s = 0; s < 2; ++s) {
            v16h aA, aB;
            ((float4*)&aA)[0] = *(const float4*)(arowA + s * 32 + hiA);
            ((float4*)&aA)[1] = *(const float4*)(arowA + s * 32 + 16 + hiA);
            ((float4*)&aB)[0] = *(const float4*)(arowB + s * 32 + hiA);
            ((float4*)&aB)[1] = *(const float4*)(arowB + s * 32 + 16 + hiA);

#define CONV_NT(CA, CB, NT)                                                     \
            {                                                                   \
                const _Float16* brow = wk + (size_t)((NT)*16 + r) * 64          \
                                          + s * 32 + hiB;                       \
                v16h bf;                                                        \
                ((float4*)&bf)[0] = ((const float4*)brow)[0];                   \
                ((float4*)&bf)[1] = ((const float4*)brow)[2];                   \
                CA = __builtin_amdgcn_wmma_f32_16x16x32_f16(                    \
                        false, aA, false, bf, (short)0, CA, false, false);      \
                CB = __builtin_amdgcn_wmma_f32_16x16x32_f16(                    \
                        false, aB, false, bf, (short)0, CB, false, false);      \
            }
            CONV_NT(accA0, accB0, 0)
            CONV_NT(accA1, accB1, 1)
            CONV_NT(accA2, accB2, 2)
            CONV_NT(accA3, accB3, 3)
#undef CONV_NT
        }
    }

    const int hsM = (lane & 16) ? 8 : 0;
#pragma unroll
    for (int v = 0; v < 8; ++v) {
        size_t rowA = (size_t)(pair * 32 + v + hsM) * 64;
        size_t rowB = rowA + (size_t)16 * 64;
        Y[rowA +      r] = accA0[v];
        Y[rowA + 16 + r] = accA1[v];
        Y[rowA + 32 + r] = accA2[v];
        Y[rowA + 48 + r] = accA3[v];
        Y[rowB +      r] = accB0[v];
        Y[rowB + 16 + r] = accB1[v];
        Y[rowB + 32 + r] = accB2[v];
        Y[rowB + 48 + r] = accB3[v];
    }
}

// ---------------------------------------------------------------------------
// per-(instance,group) sum / sumsq, LDS-staged
__global__ __launch_bounds__(256)
void k_gn_reduce(const float* __restrict__ Y, const int* __restrict__ ins,
                 float* __restrict__ gsum, float* __restrict__ gsq,
                 int npts, int pts_per_block) {
    __shared__ float ls[NUM_INST * NGROUP];
    __shared__ float lq[NUM_INST * NGROUP];
    for (int i = threadIdx.x; i < NUM_INST * NGROUP; i += 256) { ls[i] = 0.f; lq[i] = 0.f; }
    __syncthreads();
    const int g   = threadIdx.x & 31;      // group
    const int sub = threadIdx.x >> 5;      // 0..7
    const int p0  = blockIdx.x * pts_per_block;
    for (int t = sub; t < pts_per_block; t += 8) {
        int p = p0 + t;
        if (p < npts) {
            int inst = ins[p];
            float2 y = *(const float2*)(Y + (size_t)p * 64 + 2 * g);
            atomicAdd(&ls[inst * NGROUP + g], y.x + y.y);
            atomicAdd(&lq[inst * NGROUP + g], y.x * y.x + y.y * y.y);
        }
    }
    __syncthreads();
    for (int i = threadIdx.x; i < NUM_INST * NGROUP; i += 256) {
        atomicAdd(&gsum[i], ls[i]);
        atomicAdd(&gsq[i],  lq[i]);
    }
}

// ---------------------------------------------------------------------------
// mean / inv-std per (instance,group)
__global__ void k_gn_stats(const float* __restrict__ gsum, const float* __restrict__ gsq,
                           const float* __restrict__ cnt,
                           float* __restrict__ dmean, float* __restrict__ dinv) {
    int i = blockIdx.x * 256 + threadIdx.x;
    if (i < NUM_INST * NGROUP) {
        int inst = i >> 5;
        float denom = fmaxf(cnt[inst], 1.0f) * 2.0f;   // cg = 2 channels per group
        float mean = gsum[i] / denom;
        float var  = gsq[i] / denom - mean * mean;
        dmean[i] = mean;
        dinv[i]  = rsqrtf(var + EPS_GN);
    }
}

// ---------------------------------------------------------------------------
// normalize + affine + ReLU; write f16 (next-layer input) or f32 (final out)
__global__ void k_gn_norm(const float* __restrict__ Y, const int* __restrict__ ins,
                          const float* __restrict__ dmean, const float* __restrict__ dinv,
                          const float* __restrict__ gamma, const float* __restrict__ beta,
                          float* __restrict__ outF, _Float16* __restrict__ outH,
                          int n64, int writeHalf) {
    int i = blockIdx.x * 256 + threadIdx.x;
    if (i < n64) {
        int p = i >> 6;
        int c = i & 63;
        int g = c >> 1;
        int inst = ins[p];
        float v = (Y[i] - dmean[inst * NGROUP + g]) * dinv[inst * NGROUP + g]
                  * gamma[c] + beta[c];
        v = fmaxf(v, 0.0f);
        if (writeHalf) outH[i] = (_Float16)v;
        else           outF[i] = v;
    }
}

// ---------------------------------------------------------------------------
extern "C" void kernel_launch(void* const* d_in, const int* in_sizes, int n_in,
                              void* d_out, int out_size, void* d_ws, size_t ws_size,
                              hipStream_t stream) {
    const float* features = (const float*)d_in[0];
    const int*   nbr      = (const int*)  d_in[1];
    const int*   ins      = (const int*)  d_in[2];
    const float* W0  = (const float*)d_in[3];
    const float* b0  = (const float*)d_in[4];
    const float* g0  = (const float*)d_in[5];
    const float* be0 = (const float*)d_in[6];
    const float* W1  = (const float*)d_in[7];
    const float* b1  = (const float*)d_in[8];
    const float* g1  = (const float*)d_in[9];
    const float* be1 = (const float*)d_in[10];
    float* Y = (float*)d_out;                       // reuse output as conv stage buffer

    // ---- workspace carve (256B aligned) ----
    char* w = (char*)d_ws;
    size_t off = 0;
    auto take = [&](size_t bytes) { void* p = w + off; off = (off + bytes + 255) & ~(size_t)255; return p; };
    _Float16* Xh   = (_Float16*)take((size_t)(N_PTS + 1) * C_CH * sizeof(_Float16)); // ~64 MB
    _Float16* Wt   = (_Float16*)take((size_t)9 * 64 * 64 * sizeof(_Float16));
    float*    gsum = (float*)   take((size_t)NUM_INST * NGROUP * 2 * sizeof(float)); // sum+sq contiguous
    float*    gsq  = gsum + NUM_INST * NGROUP;
    float*    dmean= (float*)   take((size_t)NUM_INST * NGROUP * sizeof(float));
    float*    dinv = (float*)   take((size_t)NUM_INST * NGROUP * sizeof(float));
    float*    cnt  = (float*)   take((size_t)NUM_INST * sizeof(float));

    const int npairs = N_PTS / 32;                  // 15625 exactly
    const int n64    = N_PTS * C_CH;                // 32,000,000
    const int pts_per_block = 2048;

    // ---- instance counts (shared by both layers) ----
    k_zero_f<<<1, 256, 0, stream>>>(cnt, NUM_INST);
    k_cnt_hist<<<(N_PTS + 255) / 256, 256, 0, stream>>>(ins, cnt, N_PTS);

    // ---- features -> f16 (+ zero row) ----
    k_cvt_x<<<((N_PTS + 1) * C_CH + 255) / 256, 256, 0, stream>>>(
        features, Xh, (N_PTS + 1) * C_CH, n64);

    const float* Ws[2]  = {W0, W1};
    const float* bs[2]  = {b0, b1};
    const float* gms[2] = {g0, g1};
    const float* bes[2] = {be0, be1};

    for (int layer = 0; layer < 2; ++layer) {
        // weights -> transposed f16
        k_cvt_w<<<(9 * 64 * 64 + 255) / 256, 256, 0, stream>>>(Ws[layer], Wt);

        // sparse conv (WMMA) -> Y (= d_out buffer)
        k_conv_wmma<<<(npairs + 7) / 8, 256, 0, stream>>>(
            Xh, Wt, nbr, bs[layer], Y, npairs, N_PTS);

        // group-norm reductions
        k_zero_f<<<(NUM_INST * NGROUP * 2 + 255) / 256, 256, 0, stream>>>(
            gsum, NUM_INST * NGROUP * 2);
        k_gn_reduce<<<(N_PTS + pts_per_block - 1) / pts_per_block, 256, 0, stream>>>(
            Y, ins, gsum, gsq, N_PTS, pts_per_block);
        k_gn_stats<<<(NUM_INST * NGROUP + 255) / 256, 256, 0, stream>>>(
            gsum, gsq, cnt, dmean, dinv);

        // normalize + ReLU; layer 0 feeds Xh (f16), layer 1 writes final f32 in-place
        k_gn_norm<<<(n64 + 255) / 256, 256, 0, stream>>>(
            Y, ins, dmean, dinv, gms[layer], bes[layer],
            Y, Xh, n64, layer == 0 ? 1 : 0);
    }
    (void)in_sizes; (void)n_in; (void)out_size; (void)ws_size;
}
